// PointNet2Encoder_21921513079267
// MI455X (gfx1250) — matrix operand order repaired and checked by reading, not compiled, not run on hardware
//
#include <hip/hip_runtime.h>

// ---------------------------------------------------------------------------
// PointNet++ (MSG) encoder for MI455X / gfx1250.
// All MLP layers run as f16 WMMA GEMMs (v_wmma_f32_16x16x32_f16) with fused
// BatchNorm(+bias)+ReLU epilogue. Features are kept f16 channel-major [C][B*Np].
// Fragments are staged through LDS in fragment-major layout so each lane's
// 16 halves are contiguous -> ds_load_b128 pairs feed the matrix pipe.
// Interior blocks use unguarded vectorized tile loads (no per-element
// predication), edge blocks fall back to the guarded path.
// ---------------------------------------------------------------------------

typedef _Float16 h16;
typedef _Float16 v16h __attribute__((ext_vector_type(16)));
typedef _Float16 v8h  __attribute__((ext_vector_type(8)));
typedef float    v8f  __attribute__((ext_vector_type(8)));

union HV { v16h v; v8h h[2]; };

#define BATCH 2
#define N0 16384

// ---------------------------------------------------------------------------
// WMMA GEMM:  Y[o, m] = epi( sum_c W[o,c] * X[c,m] )
// W: f32 row-major (O x C);  X: f16 row-major (C x M)
// epi: optional +bias[o], optional BN(m,v,g,b), optional ReLU; write f16 or f32
// block: 256 threads (8 wave32), tile 32(O) x 128(M), K-chunk 32 via LDS.
// Each wave: one A fragment, two B fragments -> 2 WMMAs per chunk.
// ---------------------------------------------------------------------------
__global__ void __launch_bounds__(256)
gemm_bn_kernel(const float* __restrict__ W, const h16* __restrict__ X,
               const float* __restrict__ gamma, const float* __restrict__ beta,
               const float* __restrict__ mean,  const float* __restrict__ var,
               const float* __restrict__ bias,
               h16* __restrict__ Yh, float* __restrict__ Yf,
               int O, int C, int M, int relu)
{
    // A tile stored pre-permuted into fragment order:
    //   WtF[row*2 + khalf][i] = W[o0+row][k(i,khalf)]
    // row stride 24 halves (48B, 16B-aligned).
    __shared__ __align__(16) h16 WtF[64][24];
    // X tile transposed: XtT[m_local][k], row stride 40 halves (80B, 16B-aligned)
    __shared__ __align__(16) h16 XtT[128][40];

    const int t     = threadIdx.x;
    const int lane  = t & 31;
    const int wave  = t >> 5;
    const int wo    = wave >> 2;      // 0..1 : 16-row O sub-tile
    const int wm    = wave & 3;       // 0..3 : 32-col M sub-tile (2x16)
    const int o0    = blockIdx.y * 32;
    const int m0    = blockIdx.x * 128;
    const int khalf = lane >> 4;
    const int l15   = lane & 15;

    const bool oInt = (o0 + 32 <= O);
    const bool mInt = (m0 + 128 <= M);

    v8f acc0 = {}, acc1 = {};

    for (int k0 = 0; k0 < C; k0 += 32) {
        const bool kInt = (k0 + 32 <= C);

        // ---- W tile (f32 -> f16) in A-fragment layout ----
        if (oInt && kInt) {
            // unguarded: thread t owns elements 4t..4t+3 (same row, same k-run)
#pragma unroll
            for (int u = 0; u < 4; u++) {
                int e   = t * 4 + u;
                int row = e >> 5, t5 = e & 31;
                int kh  = t5 >> 4, i = t5 & 15;
                int k   = k0 + 16 * kh + i;
                WtF[row * 2 + kh][i] = (h16)W[(size_t)(o0 + row) * C + k];
            }
        } else {
            for (int j = t; j < 32 * 32; j += 256) {
                int row = j >> 5, t5 = j & 31;
                int kh  = t5 >> 4, i = t5 & 15;
                int k = k0 + 16 * kh + i;
                int o = o0 + row;
                WtF[row * 2 + kh][i] =
                    (o < O && k < C) ? (h16)W[(size_t)o * C + k] : (h16)0.f;
            }
        }

        // ---- X tile, transposed into LDS (global reads coalesced in m) ----
        if (mInt && kInt) {
            // unguarded: 512 runs of 8 contiguous halves (16B vector loads)
#pragma unroll
            for (int u = 0; u < 2; u++) {
                int run = t + u * 256;          // 0..511
                int r   = run >> 4;             // k row 0..31
                int c0_ = (run & 15) << 3;      // m col base 0..120
                v8h xv = *(const v8h*)&X[(size_t)(k0 + r) * M + m0 + c0_];
#pragma unroll
                for (int i = 0; i < 8; i++)
                    XtT[c0_ + i][r] = xv[i];
            }
        } else {
            for (int j = t; j < 32 * 128; j += 256) {
                int r = j >> 7, c = j & 127;
                int k = k0 + r, m = m0 + c;
                XtT[c][r] = (k < C && m < M) ? X[(size_t)k * M + m] : (h16)0.f;
            }
        }

        // prefetch next K-chunk of X into cache (global_prefetch_b8)
        {
            int pk = k0 + 32 + (t >> 6);
            if (pk < C)
                __builtin_prefetch(&X[(size_t)pk * M + m0 + ((t & 63) << 1)], 0, 1);
        }
        __syncthreads();

        // ---- build fragments with wide contiguous LDS loads ----
        HV a, b0, b1;
        const h16* ap = &WtF[(wo * 16 + l15) * 2 + khalf][0];
        a.h[0] = *(const v8h*)ap;
        a.h[1] = *(const v8h*)(ap + 8);
        const h16* bp0 = &XtT[wm * 32 + l15][16 * khalf];
        b0.h[0] = *(const v8h*)bp0;
        b0.h[1] = *(const v8h*)(bp0 + 8);
        const h16* bp1 = &XtT[wm * 32 + 16 + l15][16 * khalf];
        b1.h[0] = *(const v8h*)bp1;
        b1.h[1] = *(const v8h*)(bp1 + 8);

        acc0 = __builtin_amdgcn_wmma_f32_16x16x32_f16(
                   false, a.v, false, b0.v, (short)0, acc0, false, false);
        acc1 = __builtin_amdgcn_wmma_f32_16x16x32_f16(
                   false, a.v, false, b1.v, (short)0, acc1, false, false);
        __syncthreads();
    }

    // epilogue: D layout — VGPR r: o=r (lanes 0-15) / r+8 (lanes 16-31), n=lane%16
    const int mA = m0 + wm * 32 + l15;
    const int mB = mA + 16;
#pragma unroll
    for (int r = 0; r < 8; r++) {
        int o = o0 + wo * 16 + r + (khalf << 3);
        if (o < O) {
            float v0 = acc0[r], v1 = acc1[r];
            if (bias) { float bv = bias[o]; v0 += bv; v1 += bv; }
            if (gamma) {
                float rs = rsqrtf(var[o] + 1e-5f) * gamma[o];
                float mn = mean[o], bt = beta[o];
                v0 = (v0 - mn) * rs + bt;
                v1 = (v1 - mn) * rs + bt;
            }
            if (relu) { v0 = fmaxf(v0, 0.f); v1 = fmaxf(v1, 0.f); }
            if (Yh) {
                if (mA < M) Yh[(size_t)o * M + mA] = (h16)v0;
                if (mB < M) Yh[(size_t)o * M + mB] = (h16)v1;
            } else {
                if (mA < M) Yf[(size_t)o * M + mA] = v0;
                if (mB < M) Yf[(size_t)o * M + mB] = v1;
            }
        }
    }
}

// ---------------------------------------------------------------------------
// features (B,N,6) f32 -> f0 [6][B*N] f16
// ---------------------------------------------------------------------------
__global__ void feat_init_kernel(const float* __restrict__ feats, h16* __restrict__ f0)
{
    int g = blockIdx.x * blockDim.x + threadIdx.x;
    if (g >= BATCH * N0 * 6) return;
    int bn_ = g / 6, c = g % 6;
    f0[(size_t)c * (BATCH * N0) + bn_] = (h16)feats[g];
}

// ---------------------------------------------------------------------------
// Farthest point sampling: one 1024-thread block per batch, mind[] in regs,
// LDS argmax tree each iteration; emits sampled coordinates directly.
// ---------------------------------------------------------------------------
__global__ void __launch_bounds__(1024)
fps_kernel(const float* __restrict__ xyz, int Nsrc, int npoint, float* __restrict__ newxyz)
{
    const int b = blockIdx.x, t = threadIdx.x;
    const float* p = xyz + (size_t)b * Nsrc * 3;

    float mind[16];
    const int PT = (Nsrc + 1023) >> 10;
    for (int i = 0; i < PT; i++) mind[i] = 1e10f;

    __shared__ float sval[1024];
    __shared__ int   sidx[1024];
    __shared__ float sc[3];
    __shared__ int   sfar;

    int far = 0;
    for (int j = 0; j < npoint; j++) {
        if (t == 0) {
            sc[0] = p[far * 3 + 0]; sc[1] = p[far * 3 + 1]; sc[2] = p[far * 3 + 2];
            float* o = newxyz + ((size_t)b * npoint + j) * 3;
            o[0] = sc[0]; o[1] = sc[1]; o[2] = sc[2];
        }
        __syncthreads();
        const float cx = sc[0], cy = sc[1], cz = sc[2];
        float bv = -1.f; int bi = 0;
        for (int q = 0; q < PT; q++) {
            int n = t + (q << 10);
            if (n < Nsrc) {
                float dx = p[n * 3 + 0] - cx;
                float dy = p[n * 3 + 1] - cy;
                float dz = p[n * 3 + 2] - cz;
                float d = dx * dx + dy * dy + dz * dz;
                float mv = fminf(mind[q], d);
                mind[q] = mv;
                if (mv > bv) { bv = mv; bi = n; }
            }
        }
        sval[t] = bv; sidx[t] = bi;
        __syncthreads();
        for (int s = 512; s > 0; s >>= 1) {
            if (t < s) {
                float v2 = sval[t + s]; int i2 = sidx[t + s];
                if (v2 > sval[t] || (v2 == sval[t] && i2 < sidx[t])) {
                    sval[t] = v2; sidx[t] = i2;
                }
            }
            __syncthreads();
        }
        if (t == 0) sfar = sidx[0];
        __syncthreads();
        far = sfar;
    }
}

// ---------------------------------------------------------------------------
// Ball query: first K indices with d2 <= r2, padded with first hit.
// ---------------------------------------------------------------------------
__global__ void bq_kernel(const float* __restrict__ newxyz, const float* __restrict__ src,
                          int Nsrc, int S, int K, float r2, int* __restrict__ out)
{
    int g = blockIdx.x * blockDim.x + threadIdx.x;
    if (g >= BATCH * S) return;
    int b = g / S, s = g % S;
    const float* c = newxyz + ((size_t)b * S + s) * 3;
    const float* p = src + (size_t)b * Nsrc * 3;
    const float cx = c[0], cy = c[1], cz = c[2];
    int lidx[32]; int cnt = 0;
    for (int n = 0; n < Nsrc && cnt < K; n++) {
        float dx = p[n * 3 + 0] - cx;
        float dy = p[n * 3 + 1] - cy;
        float dz = p[n * 3 + 2] - cz;
        if (dx * dx + dy * dy + dz * dz <= r2) lidx[cnt++] = n;
    }
    int pad = (cnt > 0) ? lidx[0] : 0;
    int* o = out + (size_t)g * K;
    for (int k = 0; k < K; k++) o[k] = (k < cnt) ? lidx[k] : pad;
}

// ---------------------------------------------------------------------------
// Grouping: X[0..2][m] = centered xyz, X[3+c][m] = gathered features (f16).
// m = ((b*S+s)*K + k), X is (3+Cf) x M.
// ---------------------------------------------------------------------------
__global__ void group_kernel(const int* __restrict__ gi, const float* __restrict__ src,
                             const float* __restrict__ newxyz, const h16* __restrict__ F,
                             int Cf, int Nsrc, int S, int K, h16* __restrict__ X, int M)
{
    int m = blockIdx.x * blockDim.x + threadIdx.x;
    if (m >= M) return;
    int s = (m / K) % S;
    int b = m / (K * S);
    int n = gi[m];
    const float* p = src + ((size_t)b * Nsrc + n) * 3;
    const float* c = newxyz + ((size_t)b * S + s) * 3;
    X[(size_t)0 * M + m] = (h16)(p[0] - c[0]);
    X[(size_t)1 * M + m] = (h16)(p[1] - c[1]);
    X[(size_t)2 * M + m] = (h16)(p[2] - c[2]);
    const int ldf = BATCH * Nsrc;
    for (int cf = 0; cf < Cf; cf++)
        X[(size_t)(3 + cf) * M + m] = F[(size_t)cf * ldf + b * Nsrc + n];
}

// ---------------------------------------------------------------------------
// Max over neighborhood K:  Y (C x B*S*K) -> Fout row block at channel c0.
// ---------------------------------------------------------------------------
__global__ void maxpool_kernel(const h16* __restrict__ Y, int C, int S, int K,
                               h16* __restrict__ Fout, int c0)
{
    int g = blockIdx.x * blockDim.x + threadIdx.x;
    const int ld = BATCH * S;
    if (g >= C * ld) return;
    int c = g / ld, bs = g % ld;
    const size_t M = (size_t)ld * K;
    const h16* row = Y + (size_t)c * M + (size_t)bs * K;
    float mx = -3.0e38f;
    for (int k = 0; k < K; k++) mx = fmaxf(mx, (float)row[k]);
    Fout[(size_t)(c0 + c) * ld + bs] = (h16)mx;
}

// ---------------------------------------------------------------------------
// FP interpolation: per dense point, 3-NN in sparse set, inverse-sq-dist
// weights; writes X rows [0,C2)=interp, [C2,C2+C1)=dense skip features.
// ---------------------------------------------------------------------------
__global__ void interp_kernel(const float* __restrict__ xyzd, const float* __restrict__ xyzs,
                              int Nd, int Ns, const h16* __restrict__ Fs, int C2,
                              const h16* __restrict__ Fd, int C1, h16* __restrict__ X)
{
    int g = blockIdx.x * blockDim.x + threadIdx.x;
    const int M = BATCH * Nd;
    if (g >= M) return;
    int b = g / Nd, n = g % Nd;
    const float* pd = xyzd + ((size_t)b * Nd + n) * 3;
    const float* ps = xyzs + (size_t)b * Ns * 3;
    const float x = pd[0], y = pd[1], z = pd[2];
    float d0 = 1e30f, d1 = 1e30f, d2 = 1e30f;
    int   i0 = 0,     i1 = 0,     i2 = 0;
    for (int j = 0; j < Ns; j++) {
        float dx = ps[j * 3 + 0] - x;
        float dy = ps[j * 3 + 1] - y;
        float dz = ps[j * 3 + 2] - z;
        float d = dx * dx + dy * dy + dz * dz;
        if      (d < d0) { d2 = d1; i2 = i1; d1 = d0; i1 = i0; d0 = d; i0 = j; }
        else if (d < d1) { d2 = d1; i2 = i1; d1 = d;  i1 = j; }
        else if (d < d2) { d2 = d;  i2 = j; }
    }
    float w0 = 1.f / (fmaxf(d0, 0.f) + 1e-8f);
    float w1 = 1.f / (fmaxf(d1, 0.f) + 1e-8f);
    float w2 = 1.f / (fmaxf(d2, 0.f) + 1e-8f);
    float ws = w0 + w1 + w2;
    w0 /= ws; w1 /= ws; w2 /= ws;

    const int lds_ = BATCH * Ns;
    for (int c = 0; c < C2; c++) {
        const h16* f = Fs + (size_t)c * lds_ + b * Ns;
        float v = w0 * (float)f[i0] + w1 * (float)f[i1] + w2 * (float)f[i2];
        X[(size_t)c * M + g] = (h16)v;
    }
    const int ldd = BATCH * Nd;
    for (int c = 0; c < C1; c++)
        X[(size_t)(C2 + c) * M + g] = Fd[(size_t)c * ldd + g];
}

// ---------------------------------------------------------------------------
// Rearrange fc output [8][B*N] f32 -> d_out (B, 8, N)
// ---------------------------------------------------------------------------
__global__ void final_out_kernel(const float* __restrict__ Yf, float* __restrict__ out)
{
    int g = blockIdx.x * blockDim.x + threadIdx.x;
    if (g >= BATCH * 8 * N0) return;
    int n = g % N0;
    int o = (g / N0) % 8;
    int b = g / (N0 * 8);
    out[g] = Yf[(size_t)o * (BATCH * N0) + (size_t)b * N0 + n];
}

// ===========================================================================
// Host-side orchestration
// ===========================================================================
struct LP { const float *W, *g, *b, *m, *v; };
static inline LP lp(void* const* din, int i) {
    return LP{ (const float*)din[i],     (const float*)din[i + 1],
               (const float*)din[i + 2], (const float*)din[i + 3],
               (const float*)din[i + 4] };
}

static inline void gemm(hipStream_t st, const float* W, const h16* X,
                        const float* g, const float* bt, const float* mn, const float* vr,
                        const float* bias, h16* Yh, float* Yf,
                        int O, int C, int M, int relu)
{
    dim3 gr((M + 127) / 128, (O + 31) / 32);
    gemm_bn_kernel<<<gr, 256, 0, st>>>(W, X, g, bt, mn, vr, bias, Yh, Yf, O, C, M, relu);
}

static void sa_scale(hipStream_t st, void* const* din, int lbase,
                     const float* srcxyz, int Nsrc, const h16* Fsrc, int Cf,
                     const float* newxyz, int S, float radius, int K,
                     const int* ch, h16* Fout, int coff,
                     h16* bufA, h16* bufB, int* ballidx)
{
    const int M = BATCH * S * K;
    bq_kernel<<<(BATCH * S + 255) / 256, 256, 0, st>>>(newxyz, srcxyz, Nsrc, S, K,
                                                       radius * radius, ballidx);
    group_kernel<<<(M + 255) / 256, 256, 0, st>>>(ballidx, srcxyz, newxyz, Fsrc,
                                                  Cf, Nsrc, S, K, bufA, M);
    const int Cin = Cf + 3;
    LP p0 = lp(din, lbase), p1 = lp(din, lbase + 5), p2 = lp(din, lbase + 10);
    gemm(st, p0.W, bufA, p0.g, p0.b, p0.m, p0.v, nullptr, bufB, nullptr, ch[0], Cin,  M, 1);
    gemm(st, p1.W, bufB, p1.g, p1.b, p1.m, p1.v, nullptr, bufA, nullptr, ch[1], ch[0], M, 1);
    gemm(st, p2.W, bufA, p2.g, p2.b, p2.m, p2.v, nullptr, bufB, nullptr, ch[2], ch[1], M, 1);
    maxpool_kernel<<<(ch[2] * BATCH * S + 255) / 256, 256, 0, st>>>(bufB, ch[2], S, K, Fout, coff);
}

static void fp_stage(hipStream_t st, void* const* din, int lbase, int nlayers,
                     const float* xyzd, int Nd, const float* xyzs, int Ns,
                     const h16* Fs, int C2, const h16* Fd, int C1,
                     const int* chans, h16* Dout, h16* bufA, h16* bufB)
{
    const int M = BATCH * Nd;
    interp_kernel<<<(M + 255) / 256, 256, 0, st>>>(xyzd, xyzs, Nd, Ns, Fs, C2, Fd, C1, bufA);
    int Cin = C2 + C1;
    const h16* Xc = bufA;
    for (int l = 0; l < nlayers; l++) {
        LP p = lp(din, lbase + 5 * l);
        h16* dst = (l == nlayers - 1) ? Dout : ((Xc == bufA) ? bufB : bufA);
        gemm(st, p.W, Xc, p.g, p.b, p.m, p.v, nullptr, dst, nullptr, chans[l], Cin, M, 1);
        Xc = dst;
        Cin = chans[l];
    }
}

extern "C" void kernel_launch(void* const* d_in, const int* in_sizes, int n_in,
                              void* d_out, int out_size, void* d_ws, size_t ws_size,
                              hipStream_t stream)
{
    (void)in_sizes; (void)n_in; (void)out_size; (void)ws_size;

    const float* xyz0  = (const float*)d_in[0];
    const float* featf = (const float*)d_in[1];

    // ---- workspace carving ----
    char*  w   = (char*)d_ws;
    size_t off = 0;
    auto alloc = [&](size_t bytes) -> void* {
        void* p = w + off;
        off += (bytes + 255) & ~(size_t)255;
        return p;
    };
    h16*   f0   = (h16*)  alloc((size_t)6    * BATCH * N0    * 2);
    float* nx1  = (float*)alloc((size_t)BATCH * 1024 * 3 * 4);
    float* nx2  = (float*)alloc((size_t)BATCH * 256  * 3 * 4);
    float* nx3  = (float*)alloc((size_t)BATCH * 64   * 3 * 4);
    float* nx4  = (float*)alloc((size_t)BATCH * 16   * 3 * 4);
    h16*   f1   = (h16*)  alloc((size_t)96   * BATCH * 1024  * 2);
    h16*   f2   = (h16*)  alloc((size_t)256  * BATCH * 256   * 2);
    h16*   f3   = (h16*)  alloc((size_t)512  * BATCH * 64    * 2);
    h16*   f4   = (h16*)  alloc((size_t)1024 * BATCH * 16    * 2);
    h16*   d3b  = (h16*)  alloc((size_t)512  * BATCH * 64    * 2);
    h16*   d2b  = (h16*)  alloc((size_t)512  * BATCH * 256   * 2);
    h16*   d1b  = (h16*)  alloc((size_t)256  * BATCH * 1024  * 2);
    h16*   d0b  = (h16*)  alloc((size_t)128  * BATCH * N0    * 2);
    int*   bqid = (int*)  alloc((size_t)BATCH * 1024 * 32 * 4);
    float* fcYf = (float*)alloc((size_t)8 * BATCH * N0 * 4);
    h16*   bufA = (h16*)  alloc((size_t)262 * BATCH * N0 * 2);
    h16*   bufB = (h16*)  alloc((size_t)262 * BATCH * N0 * 2);

    // ---- input features -> f16 channel-major ----
    feat_init_kernel<<<(BATCH * N0 * 6 + 255) / 256, 256, 0, stream>>>(featf, f0);

    // ---- SA1: N=16384 -> S=1024 ----
    fps_kernel<<<BATCH, 1024, 0, stream>>>(xyz0, N0, 1024, nx1);
    {
        const int c0[3] = {16, 16, 32}, c1[3] = {32, 32, 64};
        sa_scale(stream, d_in, 2,  xyz0, N0, f0, 6, nx1, 1024, 0.05f, 16, c0, f1, 0,  bufA, bufB, bqid);
        sa_scale(stream, d_in, 17, xyz0, N0, f0, 6, nx1, 1024, 0.10f, 32, c1, f1, 32, bufA, bufB, bqid);
    }
    // ---- SA2: 1024 -> 256 ----
    fps_kernel<<<BATCH, 1024, 0, stream>>>(nx1, 1024, 256, nx2);
    {
        const int c0[3] = {64, 64, 128}, c1[3] = {64, 96, 128};
        sa_scale(stream, d_in, 32, nx1, 1024, f1, 96, nx2, 256, 0.10f, 16, c0, f2, 0,   bufA, bufB, bqid);
        sa_scale(stream, d_in, 47, nx1, 1024, f1, 96, nx2, 256, 0.20f, 32, c1, f2, 128, bufA, bufB, bqid);
    }
    // ---- SA3: 256 -> 64 ----
    fps_kernel<<<BATCH, 1024, 0, stream>>>(nx2, 256, 64, nx3);
    {
        const int c0[3] = {128, 196, 256}, c1[3] = {128, 196, 256};
        sa_scale(stream, d_in, 62, nx2, 256, f2, 256, nx3, 64, 0.20f, 16, c0, f3, 0,   bufA, bufB, bqid);
        sa_scale(stream, d_in, 77, nx2, 256, f2, 256, nx3, 64, 0.40f, 32, c1, f3, 256, bufA, bufB, bqid);
    }
    // ---- SA4: 64 -> 16 ----
    fps_kernel<<<BATCH, 1024, 0, stream>>>(nx3, 64, 16, nx4);
    {
        const int c0[3] = {256, 256, 512}, c1[3] = {256, 384, 512};
        sa_scale(stream, d_in, 92,  nx3, 64, f3, 512, nx4, 16, 0.40f, 16, c0, f4, 0,   bufA, bufB, bqid);
        sa_scale(stream, d_in, 107, nx3, 64, f3, 512, nx4, 16, 0.80f, 32, c1, f4, 512, bufA, bufB, bqid);
    }

    // ---- FP stages ----
    { const int ch[2] = {512, 512};
      fp_stage(stream, d_in, 122, 2, nx3, 64,   nx4, 16,   f4,  1024, f3, 512, ch, d3b, bufA, bufB); }
    { const int ch[2] = {512, 512};
      fp_stage(stream, d_in, 132, 2, nx2, 256,  nx3, 64,   d3b, 512,  f2, 256, ch, d2b, bufA, bufB); }
    { const int ch[2] = {256, 256};
      fp_stage(stream, d_in, 142, 2, nx1, 1024, nx2, 256,  d2b, 512,  f1, 96,  ch, d1b, bufA, bufB); }
    { const int ch[3] = {128, 128, 128};
      fp_stage(stream, d_in, 152, 3, xyz0, N0,  nx1, 1024, d1b, 256,  f0, 6,   ch, d0b, bufA, bufB); }

    // ---- FC head ----
    {
        const float* W1 = (const float*)d_in[167];
        const float* b1 = (const float*)d_in[168];
        const float* bg = (const float*)d_in[169];
        const float* bb = (const float*)d_in[170];
        const float* bm = (const float*)d_in[171];
        const float* bv = (const float*)d_in[172];
        const float* W2 = (const float*)d_in[173];
        const float* b2 = (const float*)d_in[174];
        const int M = BATCH * N0;
        // x = relu(bn(W1*f + b1))   (dropout = identity in eval)
        gemm(stream, W1, d0b, bg, bb, bm, bv, b1, bufB, nullptr, 128, 128, M, 1);
        // out = W2*x + b2 (f32)
        gemm(stream, W2, bufB, nullptr, nullptr, nullptr, nullptr, b2,
             nullptr, fcYf, 8, 128, M, 0);
        final_out_kernel<<<(BATCH * 8 * N0 + 255) / 256, 256, 0, stream>>>(fcYf, (float*)d_out);
    }
}
